// TabularTransformerWithRelPos_979252543791
// MI455X (gfx1250) — compile-verified, matrix operand order checked by reading, not run on hardware
//
#include <hip/hip_runtime.h>
#include <hip/hip_bf16.h>

// ---------------------------------------------------------------------------
// MI455X (gfx1250) implementation of TabularTransformerWithRelPos.
//  - All large GEMMs via v_wmma_f32_16x16x32_bf16 (bf16 A/B, fp32 accum).
//  - Weights converted fp32->bf16 AND pre-transposed to [N,K] once per launch
//    so hot-loop staging is uint4 -> b128 LDS stores (no guards, no scalar).
//  - Staging/epilogue addressing: uniform 64-bit base + 32-bit thread offsets
//    (global_load/store saddr form, no flat, no 64-bit VGPR pointer math).
//  - GELU/sigmoid/softmax use hardware v_exp_f32 (__expf); gelu == x*sig(2u).
//  - All tile shapes exact multiples (BM x BN templated): no bounds checks.
// ---------------------------------------------------------------------------

#define BLK_K 32
#define LDS_STRIDE 40   // halves; 80B rows: b128-aligned, conflict-free frags

typedef __attribute__((ext_vector_type(16))) __bf16 v16bf;
typedef __attribute__((ext_vector_type(8)))  float  v8f;

union Frag { v16bf v; unsigned int u[8]; };
union Acc  { v8f v; float f[8]; };

__device__ inline unsigned short f2bf(float x) {
  unsigned int u = __builtin_bit_cast(unsigned int, x);
  return (unsigned short)((u + 0x7FFFu + ((u >> 16) & 1u)) >> 16);  // RNE
}
__device__ inline unsigned int pack2(float a, float b) {
  return (unsigned int)f2bf(a) | ((unsigned int)f2bf(b) << 16);
}
__device__ inline float sigmoid_f(float x) {
  return 1.0f / (1.0f + __expf(-x));       // v_exp_f32 path
}
__device__ inline float gelu_f(float x) {
  const float c = 1.5957691216057308f;     // 2*sqrt(2/pi)
  float u = c * (x + 0.044715f * x * x * x);
  return x * sigmoid_f(u);                 // 0.5x(1+tanh(u/2)) == x*sig(u)
}

struct GemmP {
  const float* A;        // fp32 activations [.., M, K] (lda)
  const void*  B;        // bf16 weights [N, K] (pre-transposed) or fp32 acts
  const float* bias;     // [N]
  const float* resid;    // [M, N] (ldres)
  float*       C;        // [.., M, N] (ldc)
  const int*   mask;     // [B, F] int mask for score epilogues
  int K;
  int lda, ldb, ldc, ldres;
  long long sAo, sAi, sBo, sBi, sCo, sCi;  // batch strides (outer=b, inner=h)
  int nInner;            // heads per batch (>=1)
  float scale;
};

// BM/BN: block tile (BM*BN/1024 waves == 8). BTYPE: 0 fp32 B, 1 bf16 B[N,K].
// BLAYOUT (BTYPE==0 only): 0 = B is K x N (transpose-stage), 1 = B is N x K.
// EPI: 0 plain, 1 gelu, 2 sigmoid, 3 score*scale,
//      4 score*scale+relbias+key-pad, 5 score*scale+pair mask
// BIASF/RESF: compile-time bias / residual fusion.
template <int BM, int BN, int BTYPE, int BLAYOUT, int EPI, int BIASF, int RESF>
__global__ __launch_bounds__(256) void gemm_wmma_kernel(GemmP p) {
  __shared__ unsigned short sA[BM * LDS_STRIDE];
  __shared__ unsigned short sB[BN * LDS_STRIDE];

  constexpr int WN = BN / 32;            // waves along N
  constexpr int NA = BM * 8 / 256;       // A float4 units per thread
  constexpr int NB1 = BN * 4 / 256;      // B uint4 units per thread (bf16)
  constexpr int NB2 = BN * 8 / 256;      // B float4 units per thread (n-major)
  constexpr int NB3 = BN / 32;           // B float4 units per thread (k-major)

  const int t    = threadIdx.x;
  const int lane = t & 31;
  const int wave = t >> 5;
  const int wm   = wave / WN;
  const int wn   = wave % WN;
  const int half = lane >> 4;
  const int l16  = lane & 15;
  const int z    = blockIdx.z;
  const int m0   = blockIdx.y * BM;
  const int n0   = blockIdx.x * BN;
  const int zo   = z / p.nInner;
  const int zi   = z % p.nInner;

  // uniform batch-adjusted bases
  const float* Ab = p.A + (long long)zo * p.sAo + (long long)zi * p.sAi;
  const long long boff = (long long)zo * p.sBo + (long long)zi * p.sBi;
  const unsigned short* Bh = (const unsigned short*)p.B + boff;  // bf16 view
  const float* Bf = (const float*)p.B + boff;                    // fp32 view
  float* Cb = p.C + (long long)zo * p.sCo + (long long)zi * p.sCi;

  // ---- per-thread 32-bit staging offsets (stepped per K-tile) ----
  int aoff[NA]; unsigned short* asd[NA];
  for (int r = 0; r < NA; ++r) {
    int u = t + r * 256;
    int row = u >> 3, q = u & 7;
    aoff[r] = (m0 + row) * p.lda + q * 4;
    asd[r] = &sA[row * LDS_STRIDE + q * 4];
  }
  constexpr int NBF = (BLAYOUT == 1) ? NB2 : NB3;
  int b16off[NB1]; unsigned short* bsd16[NB1];
  int bfoff[NBF]; unsigned short* bsdf[NBF];
  int bstep;  // per-K-tile offset step for B
  if (BTYPE == 1) {
    bstep = BLK_K;
    for (int r = 0; r < NB1; ++r) {
      int u = t + r * 256;
      int n = u >> 2, kg = u & 3;
      b16off[r] = (n0 + n) * p.ldb + kg * 8;
      bsd16[r] = &sB[n * LDS_STRIDE + kg * 8];
    }
  } else if (BLAYOUT == 1) {
    bstep = BLK_K;
    for (int r = 0; r < NB2; ++r) {
      int u = t + r * 256;
      int n = u >> 3, q = u & 7;
      bfoff[r] = (n0 + n) * p.ldb + q * 4;
      bsdf[r] = &sB[n * LDS_STRIDE + q * 4];
    }
  } else {
    bstep = BLK_K * p.ldb;
    for (int r = 0; r < NB3; ++r) {
      int u = t + r * 256;
      int k = u / (BN / 4), nq = u % (BN / 4);
      bfoff[r] = k * p.ldb + n0 + nq * 4;
      bsdf[r] = &sB[(nq * 4) * LDS_STRIDE + k];
    }
  }

  Acc acc[2][2];
  for (int i = 0; i < 2; ++i)
    for (int j = 0; j < 2; ++j)
      for (int r = 0; r < 8; ++r) acc[i][j].f[r] = 0.0f;

  for (int k0 = 0; k0 < p.K; k0 += BLK_K) {
    // ---- stage A tile (BM x 32) ----
    for (int r = 0; r < NA; ++r) {
      float4 av = *(const float4*)(Ab + aoff[r]);
      aoff[r] += BLK_K;
      uint2 pk; pk.x = pack2(av.x, av.y); pk.y = pack2(av.z, av.w);
      *(uint2*)asd[r] = pk;
    }
    // ---- stage B tile into sB[n][k] ----
    if (BTYPE == 1) {
      for (int r = 0; r < NB1; ++r) {
        uint4 bv = *(const uint4*)(Bh + b16off[r]);
        b16off[r] += bstep;
        *(uint4*)bsd16[r] = bv;
      }
    } else if (BLAYOUT == 1) {
      for (int r = 0; r < NBF; ++r) {
        float4 bv = *(const float4*)(Bf + bfoff[r]);
        bfoff[r] += bstep;
        uint2 pk; pk.x = pack2(bv.x, bv.y); pk.y = pack2(bv.z, bv.w);
        *(uint2*)bsdf[r] = pk;
      }
    } else {
      for (int r = 0; r < NBF; ++r) {
        float4 bv = *(const float4*)(Bf + bfoff[r]);
        bfoff[r] += bstep;
        bsdf[r][0 * LDS_STRIDE] = f2bf(bv.x);
        bsdf[r][1 * LDS_STRIDE] = f2bf(bv.y);
        bsdf[r][2 * LDS_STRIDE] = f2bf(bv.z);
        bsdf[r][3 * LDS_STRIDE] = f2bf(bv.w);
      }
    }
    __syncthreads();

    // ---- fragments (ISA 7.12.2 16-bit 16x32 layout; B symmetric in N) ----
    Frag a[2], b[2];
    for (int i = 0; i < 2; ++i) {
      int row = wm * 32 + i * 16 + l16;
      for (int v = 0; v < 8; ++v) {
        int koff = (v >> 2) * 16 + half * 8 + (v & 3) * 2;
        a[i].u[v] = *(const unsigned int*)&sA[row * LDS_STRIDE + koff];
      }
    }
    for (int j = 0; j < 2; ++j) {
      int col = wn * 32 + j * 16 + l16;
      for (int v = 0; v < 8; ++v) {
        int koff = (v >> 2) * 16 + half * 8 + (v & 3) * 2;
        b[j].u[v] = *(const unsigned int*)&sB[col * LDS_STRIDE + koff];
      }
    }
    for (int i = 0; i < 2; ++i)
      for (int j = 0; j < 2; ++j)
        acc[i][j].v = __builtin_amdgcn_wmma_f32_16x16x32_bf16(
            false, a[i].v, false, b[j].v, (short)0, acc[i][j].v, false, false);
    __syncthreads();
  }

  // ---- epilogue: uniform bases + 32-bit offsets, invariants hoisted ----
  const int gmB = m0 + wm * 32;          // uniform per wave
  const int roB = half * 8;              // divergent row offset base

  int rmask[2][8];
  if (EPI == 5) {
    for (int i = 0; i < 2; ++i)
      for (int r = 0; r < 8; ++r)
        rmask[i][r] = p.mask[z * 128 + gmB + roB + i * 16 + r];
  }

  for (int j = 0; j < 2; ++j) {
    const int gn = n0 + wn * 32 + j * 16 + l16;
    float bv = 0.0f;
    if (BIASF) bv = p.bias[gn];
    int kmask = 0;
    if (EPI == 4) kmask = p.mask[zo * 128 + gn];
    if (EPI == 5) kmask = p.mask[z * 128 + gn];
    for (int i = 0; i < 2; ++i)
      for (int r = 0; r < 8; ++r) {
        const int ro = roB + i * 16 + r;           // row offset from gmB
        float v = acc[i][j].f[r];
        if (EPI <= 2) {
          if (BIASF) v += bv;
          if (EPI == 1) v = gelu_f(v);
          else if (EPI == 2) v = sigmoid_f(v);
          if (RESF) v += p.resid[(gmB + ro) * p.ldres + gn];
        } else if (EPI == 3) {
          v *= p.scale;
        } else if (EPI == 4) {  // rel-pos bias + key padding
          v = kmask ? -1e9f
                    : (v * p.scale - 0.1f * fabsf((float)(gmB + ro - gn)));
        } else {                // EPI == 5: FCM observed-pair mask
          v = (rmask[i][r] == 0 && kmask == 0) ? v * p.scale : -1e9f;
        }
        Cb[(gmB + ro) * p.ldc + gn] = v;
      }
  }
}

// ---------------- elementwise / reduction kernels ----------------

// fp32 [K,N] -> bf16 [N,K] (weights; once per launch). N must be a power of 2.
__global__ void cvtT_bf16_kernel(const float* src, unsigned short* dst,
                                 int K, int nshift) {
  int total = K << nshift;
  int nmask = (1 << nshift) - 1;
  for (int i = blockIdx.x * blockDim.x + threadIdx.x; i < total;
       i += gridDim.x * blockDim.x) {
    int k = i >> nshift, n = i & nmask;
    dst[n * K + k] = f2bf(src[i]);
  }
}

__global__ void embed_kernel(const float* x, const float* in_w, const float* in_b,
                             const float* fe, float* h, long long total) {
  for (long long i = (long long)blockIdx.x * blockDim.x + threadIdx.x; i < total;
       i += (long long)gridDim.x * blockDim.x) {
    int d = (int)(i & 511);
    long long m = i >> 9;
    int f = (int)(m & 127);
    h[i] = x[m] * in_w[d] + in_b[d] + fe[f * 512 + d];
  }
}

__global__ void me1_kernel(const int* mask, const float* w1, const float* b1,
                           float* out, int M) {
  int total = M * 256;
  for (int i = blockIdx.x * blockDim.x + threadIdx.x; i < total;
       i += gridDim.x * blockDim.x) {
    int m = i >> 8, j = i & 255;
    out[i] = gelu_f((float)mask[m] * w1[j] + b1[j]);
  }
}

// wave-per-row LayerNorm over D=512. GATED: in = a + g*c
template <int GATED>
__global__ __launch_bounds__(256) void ln_kernel(const float* a, const float* g,
                                                 const float* c, const float* gamma,
                                                 const float* beta, float* out,
                                                 int ldin, int ldout, int M) {
  int wave = threadIdx.x >> 5, lane = threadIdx.x & 31;
  long long row = (long long)blockIdx.x * 8 + wave;
  if (row >= M) return;
  const float* pa = a + row * ldin;
  float vals[16];
  for (int i = 0; i < 4; ++i) {
    float4 va = ((const float4*)pa)[lane * 4 + i];
    if (GATED) {
      float4 vg = ((const float4*)(g + row * 512))[lane * 4 + i];
      float4 vc = ((const float4*)(c + row * 512))[lane * 4 + i];
      va.x += vg.x * vc.x; va.y += vg.y * vc.y;
      va.z += vg.z * vc.z; va.w += vg.w * vc.w;
    }
    vals[i * 4 + 0] = va.x; vals[i * 4 + 1] = va.y;
    vals[i * 4 + 2] = va.z; vals[i * 4 + 3] = va.w;
  }
  float s = 0.f, q = 0.f;
  for (int i = 0; i < 16; ++i) { s += vals[i]; q += vals[i] * vals[i]; }
  for (int m = 16; m >= 1; m >>= 1) { s += __shfl_xor(s, m); q += __shfl_xor(q, m); }
  float mean = s * (1.0f / 512.0f);
  float var  = q * (1.0f / 512.0f) - mean * mean;
  float inv  = rsqrtf(var + 1e-5f);
  float* po = out + row * ldout;
  for (int i = 0; i < 16; ++i) {
    int col = lane * 16 + i;
    po[col] = (vals[i] - mean) * inv * gamma[col] + beta[col];
  }
}

// wave-per-row softmax over 128 columns (hardware exp)
__global__ __launch_bounds__(256) void softmax128_kernel(float* S, long long rows) {
  int wave = threadIdx.x >> 5, lane = threadIdx.x & 31;
  long long row = (long long)blockIdx.x * 8 + wave;
  if (row >= rows) return;
  float4* p = (float4*)(S + row * 128);
  float4 v = p[lane];
  float mx = fmaxf(fmaxf(v.x, v.y), fmaxf(v.z, v.w));
  for (int m = 16; m >= 1; m >>= 1) mx = fmaxf(mx, __shfl_xor(mx, m));
  v.x = __expf(v.x - mx); v.y = __expf(v.y - mx);
  v.z = __expf(v.z - mx); v.w = __expf(v.w - mx);
  float s = v.x + v.y + v.z + v.w;
  for (int m = 16; m >= 1; m >>= 1) s += __shfl_xor(s, m);
  float inv = 1.0f / s;
  v.x *= inv; v.y *= inv; v.z *= inv; v.w *= inv;
  p[lane] = v;
}

// wave-per-row final projection D -> 1
__global__ __launch_bounds__(256) void out_kernel(const float* h, const float* w,
                                                  const float* b, float* out, int M) {
  int wave = threadIdx.x >> 5, lane = threadIdx.x & 31;
  long long row = (long long)blockIdx.x * 8 + wave;
  if (row >= M) return;
  const float* ph = h + row * 512;
  float s = 0.f;
  for (int i = 0; i < 4; ++i) {
    float4 a = ((const float4*)ph)[lane * 4 + i];
    float4 ww = ((const float4*)w)[lane * 4 + i];
    s += a.x * ww.x + a.y * ww.y + a.z * ww.z + a.w * ww.w;
  }
  for (int m = 16; m >= 1; m >>= 1) s += __shfl_xor(s, m);
  if (lane == 0) out[row] = s + b[0];
}

// ---------------------------------------------------------------------------

extern "C" void kernel_launch(void* const* d_in, const int* in_sizes, int n_in,
                              void* d_out, int out_size, void* d_ws, size_t ws_size,
                              hipStream_t stream) {
  (void)in_sizes; (void)n_in; (void)out_size; (void)ws_size;
  const int Bb = 256, F = 128, D = 512, H = 8, HF = 4, HD = 64, HDF = 128,
            FFN = 2048, NL = 4;
  const int M = Bb * F;                       // 32768
  const long long MD = (long long)M * D;      // 16,777,216

  const float* X     = (const float*)d_in[0];
  const int*   Imask = (const int*)d_in[1];
  auto f32 = [&](int i) { return (const float*)d_in[i]; };

  // ---- workspace layout ----
  unsigned short* wb = (unsigned short*)d_ws;   // bf16 weight arena (16 MB)
  float* fb = (float*)((char*)d_ws + (16ll << 20));
  float* buf0  = fb + 0 * MD;
  float* buf1  = fb + 1 * MD;
  float* buf2  = fb + 2 * MD;
  float* buf3  = fb + 3 * MD;
  float* buf4  = fb + 4 * MD;
  float* buf5  = fb + 5 * MD;
  float* comb  = fb + 6 * MD;                   // [M, 2D]
  float* bufFF = fb + 8 * MD;                   // [M, 2048]; aliases score buf

  // ---- convert + transpose weights into bf16 [N,K] arena ----
  size_t wcur = 0;
  auto cvtW = [&](int idx, int K, int N) -> const unsigned short* {
    unsigned short* dst = wb + wcur; wcur += (size_t)K * N;
    int nshift = (N == 2048) ? 11 : 9;          // N in {512, 2048}
    int n = K * N;
    int blocks = (n + 1023) / 1024; if (blocks > 8192) blocks = 8192;
    cvtT_bf16_kernel<<<blocks, 256, 0, stream>>>(f32(idx), dst, K, nshift);
    return dst;
  };
  const unsigned short* w_ve1 = cvtW(5, D, D);
  const unsigned short* w_ve2 = cvtW(7, D, D);
  const unsigned short* w_ve3 = cvtW(9, D, D);
  const unsigned short* w_me2 = cvtW(15, 256, D);
  const unsigned short* w_fq  = cvtW(19, D, D);
  const unsigned short* w_fk  = cvtW(21, D, D);
  const unsigned short* w_fv  = cvtW(23, D, D);
  const unsigned short* w_fo  = cvtW(25, D, D);
  const unsigned short* w_fu1 = cvtW(27, 2 * D, D);
  const unsigned short* w_fu2 = cvtW(29, D, D);
  const unsigned short* w_cp  = cvtW(33, D, D);
  const unsigned short* w_g   = cvtW(35, D, D);
  const unsigned short* w_lq  = cvtW(39, D, D);
  const unsigned short* w_lk  = cvtW(41, D, D);
  const unsigned short* w_lv  = cvtW(43, D, D);
  const unsigned short* w_lo  = cvtW(45, D, D);
  const unsigned short* w_l1  = cvtW(47, D, FFN);
  const unsigned short* w_l2  = cvtW(49, FFN, D);

  // ---- launch helpers ----
  auto gemmW = [&](const float* A, int lda, const unsigned short* Bw,
                   const float* bias, const float* resid, int ldres, float* C,
                   int ldc, int Mm, int Nn, int Kk, int act) {
    GemmP p{}; p.A = A; p.B = Bw; p.bias = bias; p.resid = resid; p.C = C;
    p.mask = nullptr; p.K = Kk;
    p.lda = lda; p.ldb = Kk; p.ldc = ldc; p.ldres = ldres;  // ldb = K ([N,K])
    p.nInner = 1; p.scale = 1.0f;
    dim3 g(Nn / 128, Mm / 64, 1);
    if (act == 1)
      gemm_wmma_kernel<64, 128, 1, 1, 1, 1, 0><<<g, 256, 0, stream>>>(p);
    else if (act == 2)
      gemm_wmma_kernel<64, 128, 1, 1, 2, 1, 0><<<g, 256, 0, stream>>>(p);
    else if (resid)
      gemm_wmma_kernel<64, 128, 1, 1, 0, 1, 1><<<g, 256, 0, stream>>>(p);
    else
      gemm_wmma_kernel<64, 128, 1, 1, 0, 1, 0><<<g, 256, 0, stream>>>(p);
  };
  auto softmax = [&](float* S, long long rows) {
    softmax128_kernel<<<(int)((rows + 7) / 8), 256, 0, stream>>>(S, rows);
  };
  auto lnrm = [&](const float* a, const float* gam, const float* bet, float* o,
                  int ldin, int ldout) {
    ln_kernel<0><<<M / 8, 256, 0, stream>>>(a, nullptr, nullptr, gam, bet, o,
                                            ldin, ldout, M);
  };

  // ================= embedding =================
  embed_kernel<<<8192, 256, 0, stream>>>(X, f32(2), f32(3), f32(4), buf0, MD);

  // ================= FVDE =================
  gemmW(buf0, D, w_ve1, f32(6), nullptr, 0, buf1, D, M, D, D, 1);      // gelu
  gemmW(buf1, D, w_ve2, f32(8), nullptr, 0, buf2, D, M, D, D, 1);      // gelu
  gemmW(buf2, D, w_ve3, f32(10), nullptr, 0, buf1, D, M, D, D, 0);
  lnrm(buf1, f32(11), f32(12), comb, D, 2 * D);                        // ve

  me1_kernel<<<8192, 256, 0, stream>>>(Imask, f32(13), f32(14), buf2, M);
  gemmW(buf2, 256, w_me2, f32(16), nullptr, 0, buf3, D, M, D, 256, 0);
  lnrm(buf3, f32(17), f32(18), buf4, D, D);                            // meL

  gemmW(buf4, D, w_fq, f32(20), nullptr, 0, buf0, D, M, D, D, 0);      // q
  gemmW(comb, 2 * D, w_fk, f32(22), nullptr, 0, buf1, D, M, D, D, 0);  // k
  gemmW(comb, 2 * D, w_fv, f32(24), nullptr, 0, buf2, D, M, D, D, 0);  // v

  {  // scores: [B,HF,F,F] = Q K^T / sqrt(HDF)
    GemmP p{}; p.A = buf0; p.B = buf1; p.C = bufFF;
    p.K = HDF; p.lda = D; p.ldb = D; p.ldc = F;
    p.sAo = (long long)F * D; p.sAi = HDF;
    p.sBo = (long long)F * D; p.sBi = HDF;
    p.sCo = (long long)HF * F * F; p.sCi = (long long)F * F;
    p.nInner = HF; p.scale = 0.08838834764831845f;  // 1/sqrt(128)
    dim3 g(1, 2, Bb * HF);
    gemm_wmma_kernel<64, 128, 0, 1, 3, 0, 0><<<g, 256, 0, stream>>>(p);
  }
  softmax(bufFF, (long long)Bb * HF * F);
  {  // ao = P V -> buf3 [B,F,HF,HDF]
    GemmP p{}; p.A = bufFF; p.B = buf2; p.C = buf3;
    p.K = F; p.lda = F; p.ldb = D; p.ldc = D;
    p.sAo = (long long)HF * F * F; p.sAi = (long long)F * F;
    p.sBo = (long long)F * D; p.sBi = HDF;
    p.sCo = (long long)F * D; p.sCi = HDF;
    p.nInner = HF; p.scale = 1.0f;
    dim3 g(1, 2, Bb * HF);
    gemm_wmma_kernel<64, 128, 0, 0, 0, 0, 0><<<g, 256, 0, stream>>>(p);
  }
  gemmW(buf3, D, w_fo, f32(26), buf4, D, comb + D, 2 * D, M, D, D, 0); // me
  gemmW(comb, 2 * D, w_fu1, f32(28), nullptr, 0, buf0, D, M, D, 2 * D, 1);
  gemmW(buf0, D, w_fu2, f32(30), nullptr, 0, buf1, D, M, D, D, 0);
  lnrm(buf1, f32(31), f32(32), buf5, D, D);                            // enh

  // ================= FCM =================
  gemmW(buf5, D, w_cp, f32(34), nullptr, 0, buf0, D, M, D, D, 0);      // xp
  {  // corr = xp xp^T / sqrt(D) with observed-pair mask
    GemmP p{}; p.A = buf0; p.B = buf0; p.C = bufFF; p.mask = Imask;
    p.K = D; p.lda = D; p.ldb = D; p.ldc = F;
    p.sAo = (long long)F * D; p.sBo = (long long)F * D;
    p.sCo = (long long)F * F;
    p.nInner = 1; p.scale = 0.044194173824159216f;  // 1/sqrt(512)
    dim3 g(1, 2, Bb);
    gemm_wmma_kernel<64, 128, 0, 1, 5, 0, 0><<<g, 256, 0, stream>>>(p);
  }
  softmax(bufFF, (long long)Bb * F);
  {  // cf = cw @ enh
    GemmP p{}; p.A = bufFF; p.B = buf5; p.C = buf1;
    p.K = F; p.lda = F; p.ldb = D; p.ldc = D;
    p.sAo = (long long)F * F; p.sBo = (long long)F * D;
    p.sCo = (long long)F * D; p.nInner = 1; p.scale = 1.0f;
    dim3 g(4, 2, Bb);
    gemm_wmma_kernel<64, 128, 0, 0, 0, 0, 0><<<g, 256, 0, stream>>>(p);
  }
  gemmW(buf5, D, w_g, f32(36), nullptr, 0, buf2, D, M, D, D, 2);       // gates
  ln_kernel<1><<<M / 8, 256, 0, stream>>>(buf5, buf2, buf1, f32(37), f32(38),
                                          buf0, D, D, M);              // h

  // ================= transformer layers (shared weights) =================
  for (int l = 0; l < NL; ++l) {
    lnrm(buf0, f32(51), f32(52), buf1, D, D);                          // s2
    gemmW(buf1, D, w_lq, f32(40), nullptr, 0, buf2, D, M, D, D, 0);    // q
    gemmW(buf1, D, w_lk, f32(42), nullptr, 0, buf3, D, M, D, D, 0);    // k
    gemmW(buf1, D, w_lv, f32(44), nullptr, 0, buf4, D, M, D, D, 0);    // v
    {  // scores with rel-pos bias + key padding mask
      GemmP p{}; p.A = buf2; p.B = buf3; p.C = bufFF; p.mask = Imask;
      p.K = HD; p.lda = D; p.ldb = D; p.ldc = F;
      p.sAo = (long long)F * D; p.sAi = HD;
      p.sBo = (long long)F * D; p.sBi = HD;
      p.sCo = (long long)H * F * F; p.sCi = (long long)F * F;
      p.nInner = H; p.scale = 0.125f;  // 1/sqrt(64)
      dim3 g(1, 2, Bb * H);
      gemm_wmma_kernel<64, 128, 0, 1, 4, 0, 0><<<g, 256, 0, stream>>>(p);
    }
    softmax(bufFF, (long long)Bb * H * F);
    {  // o = P V -> buf1 [B,F,H,HD]   (BM=128, BN=64 -> full tile)
      GemmP p{}; p.A = bufFF; p.B = buf4; p.C = buf1;
      p.K = F; p.lda = F; p.ldb = D; p.ldc = D;
      p.sAo = (long long)H * F * F; p.sAi = (long long)F * F;
      p.sBo = (long long)F * D; p.sBi = HD;
      p.sCo = (long long)F * D; p.sCi = HD;
      p.nInner = H; p.scale = 1.0f;
      dim3 g(1, 1, Bb * H);
      gemm_wmma_kernel<128, 64, 0, 0, 0, 0, 0><<<g, 256, 0, stream>>>(p);
    }
    gemmW(buf1, D, w_lo, f32(46), buf0, D, buf0, D, M, D, D, 0);       // resid
    lnrm(buf0, f32(53), f32(54), buf1, D, D);                          // s2
    gemmW(buf1, D, w_l1, f32(48), nullptr, 0, bufFF, FFN, M, FFN, D, 1);
    gemmW(bufFF, FFN, w_l2, f32(50), buf0, D, buf0, D, M, D, FFN, 0);
  }

  // ================= output projection =================
  out_kernel<<<M / 8, 256, 0, stream>>>(buf0, f32(55), f32(56), (float*)d_out, M);
}
